// CVRPModel_50697793962833
// MI455X (gfx1250) — matrix operand order repaired and checked by reading, not compiled, not run on hardware
//
#include <hip/hip_runtime.h>
#include <math.h>

// ---------------- problem constants ----------------
#define Bc 16      // batch
#define NN 501     // N+1 nodes
#define NP 512     // padded nodes
#define Ec 128     // embedding
#define Hc 8       // heads
#define Dc 16      // head dim
#define Fc 512     // ff hidden
#define Lc 6       // encoder layers
#define Pc 100     // pomo
#define PPc 112    // padded pomo (7 * 16)

typedef __bf16 bf16_t;
typedef __attribute__((ext_vector_type(16))) __bf16       v16bf;
typedef __attribute__((ext_vector_type(8)))  float        v8f;
typedef __attribute__((ext_vector_type(4)))  float        v4f;
typedef __attribute__((ext_vector_type(4)))  unsigned int v4u;
typedef __attribute__((ext_vector_type(2)))  unsigned int v2u;

union frag16 { v16bf v; v4u q[2]; };

// ---------------- bf16 helpers ----------------
__device__ __forceinline__ bf16_t bfbits(unsigned short u) {
  union { unsigned short s; bf16_t b; } c; c.s = u; return c.b;
}
__device__ __forceinline__ bf16_t f2bf(float f) {
  unsigned int u = __float_as_uint(f);
  u += 0x7FFFu + ((u >> 16) & 1u);          // round to nearest even
  return bfbits((unsigned short)(u >> 16));
}

// ---------------- WMMA fragment loaders (CDNA5 wave32 layouts) ----------------
// A 16x32 tile: lane m=l&15,h=l>>4 holds elems [k0+h*8,+8) and [k0+16+h*8,+8).
__device__ __forceinline__ v16bf load_a_al(const bf16_t* A, int lda, int row0, int k0,
                                           int lane) {
  int m = lane & 15, h = lane >> 4;
  const bf16_t* p = A + (size_t)(row0 + m) * lda + k0 + h * 8;
  frag16 f;
  f.q[0] = *(const v4u*)p;
  f.q[1] = *(const v4u*)(p + 16);
  return f.v;
}
// A tile with only K=16 valid (k0=0): high half zero.
__device__ __forceinline__ v16bf load_a16(const bf16_t* A, int lda, int row0, int lane) {
  int m = lane & 15, h = lane >> 4;
  const bf16_t* p = A + (size_t)(row0 + m) * lda + h * 8;
  frag16 f;
  f.q[0] = *(const v4u*)p;
  f.q[1] = (v4u){0u, 0u, 0u, 0u};
  return f.v;
}
// B-transposed source: frag[k][n] = S[n0+n][k] -> elems [k0+h*8,+8) and [k0+16+h*8,+8).
__device__ __forceinline__ v16bf load_bt_al(const bf16_t* S, int lds, int k0, int n0,
                                            int lane) {
  int n = lane & 15, h = lane >> 4;
  const bf16_t* p = S + (size_t)(n0 + n) * lds + k0 + h * 8;
  frag16 f;
  f.q[0] = *(const v4u*)p;
  f.q[1] = *(const v4u*)(p + 16);
  return f.v;
}
__device__ __forceinline__ v8f wmma_bf16(v16bf a, v16bf b, v8f c) {
  return __builtin_amdgcn_wmma_f32_16x16x32_bf16(false, a, false, b, (short)0, c,
                                                 false, false);
}

// ---------------- weight pack: f32 [K,N] -> bf16 B-fragment order, nt-major ----------
// dst[((nt*Ktiles + kt)*32 + lane)*16 + e] = W[kt*32 + (lane>>4)*8 + (e&7) + (e>>3)*16][nt*16 + (lane&15)]
// grid(ceil(K/32), N/16), block(16,32). Per-K-step stride = 512 elems (constant 1KB).
__global__ void pack_b_kernel(const float* __restrict__ src, bf16_t* __restrict__ dst,
                              int K, int N) {
  int e = threadIdx.x;       // 0..15
  int lane = threadIdx.y;    // 0..31
  int kt = blockIdx.x, nt = blockIdx.y;
  int Ktiles = gridDim.x;
  int k = kt * 32 + ((lane >> 4) << 3) + (e & 7) + ((e >> 3) << 4);
  int n = nt * 16 + (lane & 15);
  float v = (k < K) ? src[(size_t)k * N + n] : 0.0f;
  dst[(((size_t)nt * Ktiles + kt) * 32 + lane) * 16 + e] = f2bf(v);
}

// ---------------- embedding ----------------
// grid (NP, Bc), block Ec
__global__ void embed_kernel(const float* __restrict__ depot_xy,
                             const float* __restrict__ node_xy,
                             const float* __restrict__ node_demand,
                             const float* __restrict__ edw, const float* __restrict__ edb,
                             const float* __restrict__ enw, const float* __restrict__ enb,
                             float* __restrict__ xf, bf16_t* __restrict__ xb) {
  int n = blockIdx.x, b = blockIdx.y, e = threadIdx.x;
  float v = 0.0f;
  if (n == 0) {
    v = depot_xy[b * 2 + 0] * edw[e] + depot_xy[b * 2 + 1] * edw[Ec + e] + edb[e];
  } else if (n < NN) {
    int i = n - 1;
    v = node_xy[((size_t)b * 500 + i) * 2 + 0] * enw[e] +
        node_xy[((size_t)b * 500 + i) * 2 + 1] * enw[Ec + e] +
        node_demand[(size_t)b * 500 + i] * enw[2 * Ec + e] + enb[e];
  }
  size_t idx = ((size_t)b * NP + n) * Ec + e;
  xf[idx] = v;
  xb[idx] = f2bf(v);
}

// ---------------- generic bf16 WMMA GEMM: 16 rows x 64 cols per wave ----------------
// grid(Mtiles, ceil((Ntiles/4)/4)), block(32,4); K multiple of 32 (zero-padded);
// Ntiles multiple of 4.
__global__ void gemm_kernel(const bf16_t* __restrict__ A, int lda,
                            const bf16_t* __restrict__ Wp,
                            int K, int Ntiles,
                            const float* __restrict__ bias,
                            const float* __restrict__ residual, int ldr,
                            float* __restrict__ outf, bf16_t* __restrict__ outb, int ldo,
                            bf16_t* __restrict__ outbT, int tShift, int tCols,
                            float scale, int relu) {
  int lane = threadIdx.x;
  int ng = blockIdx.y * 4 + threadIdx.y;      // column group of 4 tiles
  int colGroups = Ntiles >> 2;
  if (ng >= colGroups) return;
  int nt0 = ng * 4;
  int row0 = blockIdx.x * 16;
  int Ktiles = K >> 5;
  int m = lane & 15, h = lane >> 4;
  const bf16_t* ap  = A + (size_t)(row0 + m) * lda + h * 8;
  size_t wslice = (size_t)Ktiles * 512;       // elems per nt block
  const bf16_t* bp0 = Wp + (size_t)nt0 * wslice + (size_t)lane * 16;
  const bf16_t* bp1 = bp0 + wslice;
  const bf16_t* bp2 = bp1 + wslice;
  const bf16_t* bp3 = bp2 + wslice;
  v8f ac0 = {}, ac1 = {}, ac2 = {}, ac3 = {};
  for (int ks = 0; ks < Ktiles; ++ks) {
    frag16 a;
    a.q[0] = *(const v4u*)(ap);
    a.q[1] = *(const v4u*)(ap + 16);
    ap += 32;
    frag16 b0, b1, b2, b3;
    b0.q[0] = *(const v4u*)(bp0); b0.q[1] = *(const v4u*)(bp0 + 8); bp0 += 512;
    b1.q[0] = *(const v4u*)(bp1); b1.q[1] = *(const v4u*)(bp1 + 8); bp1 += 512;
    b2.q[0] = *(const v4u*)(bp2); b2.q[1] = *(const v4u*)(bp2 + 8); bp2 += 512;
    b3.q[0] = *(const v4u*)(bp3); b3.q[1] = *(const v4u*)(bp3 + 8); bp3 += 512;
    ac0 = wmma_bf16(a.v, b0.v, ac0);
    ac1 = wmma_bf16(a.v, b1.v, ac1);
    ac2 = wmma_bf16(a.v, b2.v, ac2);
    ac3 = wmma_bf16(a.v, b3.v, ac3);
  }
  v8f accv[4] = {ac0, ac1, ac2, ac3};
  #pragma unroll
  for (int t = 0; t < 4; ++t) {
    int n = (nt0 + t) * 16 + (lane & 15);
    float bv = bias ? bias[n] : 0.0f;
    #pragma unroll
    for (int r = 0; r < 8; ++r) {
      int row = row0 + ((lane >> 4) << 3) + r;
      float v = accv[t][r] * scale + bv;
      if (relu) v = fmaxf(v, 0.0f);
      if (residual) v += residual[(size_t)row * ldr + n];
      if (outf) outf[(size_t)row * ldo + n] = v;
      if (outb) outb[(size_t)row * ldo + n] = f2bf(v);
      if (outbT) {  // transposed per-batch store: [b][n][localRow]
        int bb = row >> tShift;
        int lr = row - (bb << tShift);
        outbT[(((size_t)bb * tCols + n) << tShift) + lr] = f2bf(v);
      }
    }
  }
}

// ---------------- transposed-B GEMM (q_ref @ encoded^T), 4 tiles/wave --------------
// grid(Mtiles, Ntiles/4, B), block 32
__global__ void gemm_bt_kernel(const bf16_t* __restrict__ A, int lda, int apitch,
                               const bf16_t* __restrict__ Bsrc, int ldb, int bpitch,
                               float* __restrict__ outf, int ldo, int opitch,
                               int K, float scale) {
  int lane = threadIdx.x;
  int mt = blockIdx.x, nt0 = blockIdx.y * 4, b = blockIdx.z;
  int m = lane & 15, h = lane >> 4;
  const bf16_t* ap = A + (size_t)b * apitch + (size_t)(mt * 16 + m) * lda + h * 16;
  const bf16_t* Bb = Bsrc + (size_t)b * bpitch + h * 16;
  const bf16_t* bq0 = Bb + (size_t)((nt0 + 0) * 16 + m) * ldb;
  const bf16_t* bq1 = Bb + (size_t)((nt0 + 1) * 16 + m) * ldb;
  const bf16_t* bq2 = Bb + (size_t)((nt0 + 2) * 16 + m) * ldb;
  const bf16_t* bq3 = Bb + (size_t)((nt0 + 3) * 16 + m) * ldb;
  v8f ac0 = {}, ac1 = {}, ac2 = {}, ac3 = {};
  int Ktiles = K >> 5;
  for (int ks = 0; ks < Ktiles; ++ks) {
    frag16 a;
    a.q[0] = *(const v4u*)(ap);
    a.q[1] = *(const v4u*)(ap + 8);
    ap += 32;
    frag16 b0, b1, b2, b3;
    b0.q[0] = *(const v4u*)(bq0); b0.q[1] = *(const v4u*)(bq0 + 8); bq0 += 32;
    b1.q[0] = *(const v4u*)(bq1); b1.q[1] = *(const v4u*)(bq1 + 8); bq1 += 32;
    b2.q[0] = *(const v4u*)(bq2); b2.q[1] = *(const v4u*)(bq2 + 8); bq2 += 32;
    b3.q[0] = *(const v4u*)(bq3); b3.q[1] = *(const v4u*)(bq3 + 8); bq3 += 32;
    ac0 = wmma_bf16(a.v, b0.v, ac0);
    ac1 = wmma_bf16(a.v, b1.v, ac1);
    ac2 = wmma_bf16(a.v, b2.v, ac2);
    ac3 = wmma_bf16(a.v, b3.v, ac3);
  }
  v8f accv[4] = {ac0, ac1, ac2, ac3};
  float* ob = outf + (size_t)b * opitch;
  #pragma unroll
  for (int t = 0; t < 4; ++t) {
    int n = (nt0 + t) * 16 + (lane & 15);
    #pragma unroll
    for (int r = 0; r < 8; ++r) {
      int row = mt * 16 + ((lane >> 4) << 3) + r;
      ob[(size_t)row * ldo + n] = accv[t][r] * scale;
    }
  }
}

// ---------------- fused MHA (scores -> softmax -> @V) ----------------
// grid(rowTiles, H, B), block 32 (single wave).
// q:[B,qpitch,128], kc:[B,NP,128], vt:[B,128,NP] (transposed V)
__global__ void attn_kernel(const bf16_t* __restrict__ q, int qpitch,
                            const bf16_t* __restrict__ kc,
                            const bf16_t* __restrict__ vt,
                            const float* __restrict__ mask, int maskP, int maskN,
                            bf16_t* __restrict__ out,
                            int mvalid, float scale) {
  __shared__ __align__(16) float  S[16 * NP];
  __shared__ __align__(16) bf16_t Wl[16 * NP];
  int lane = threadIdx.x;
  int nt = blockIdx.x, h = blockIdx.y, b = blockIdx.z;
  int n0 = nt * 16;
  int hh = lane >> 4, nn = lane & 15;
  // Q fragment: 16 rows x 16 d (upper K half zero)
  v16bf qf = load_a16(q + (size_t)b * qpitch * Ec + h * Dc, Ec, n0, lane);
  // K rows: loop-invariant per-lane base + constant per-tile offset
  const bf16_t* kp = kc + (size_t)b * NP * Ec + h * Dc + (size_t)nn * Ec;
  v4u z = {0u, 0u, 0u, 0u};
  for (int mt = 0; mt < NP / 16; ++mt) {
    const bf16_t* p = kp + (size_t)mt * (16 * Ec);
    v4u lo = *(const v4u*)p;
    v4u hi = *(const v4u*)(p + 8);
    frag16 fb;
    fb.q[0] = hh ? hi : lo;
    fb.q[1] = z;
    v8f acc = {};
    acc = wmma_bf16(qf, fb.v, acc);
    int col = mt * 16 + nn;
    #pragma unroll
    for (int r = 0; r < 8; ++r) {
      int row = (hh << 3) + r;
      float v = acc[r] * scale;
      if (col >= mvalid) {
        v = -1e30f;
      } else if (mask) {
        int prow = n0 + row;
        if (prow < maskP) v += mask[((size_t)b * maskP + prow) * maskN + col];
      }
      S[row * NP + col] = v;
    }
  }
  __syncthreads();
  // softmax: lane = (row = l&15, half = l>>4); halves combined via shfl_xor(16)
  int rbase = nn * NP + hh * (NP / 2);
  float mx = -1e30f;
  for (int c = 0; c < NP / 2; c += 4) {
    v4f sv = *(const v4f*)(&S[rbase + c]);
    mx = fmaxf(mx, fmaxf(fmaxf(sv.x, sv.y), fmaxf(sv.z, sv.w)));
  }
  mx = fmaxf(mx, __shfl_xor(mx, 16, 32));
  float sum = 0.0f;
  for (int c = 0; c < NP / 2; c += 4) {
    v4f sv = *(const v4f*)(&S[rbase + c]);
    sv.x = __expf(sv.x - mx); sv.y = __expf(sv.y - mx);
    sv.z = __expf(sv.z - mx); sv.w = __expf(sv.w - mx);
    *(v4f*)(&S[rbase + c]) = sv;
    sum += (sv.x + sv.y) + (sv.z + sv.w);
  }
  sum += __shfl_xor(sum, 16, 32);
  float inv = 1.0f / sum;
  for (int c = 0; c < NP / 2; c += 4) {
    v4f sv = *(const v4f*)(&S[rbase + c]);
    union { v2u u; bf16_t b[4]; } pk;
    pk.b[0] = f2bf(sv.x * inv); pk.b[1] = f2bf(sv.y * inv);
    pk.b[2] = f2bf(sv.z * inv); pk.b[3] = f2bf(sv.w * inv);
    *(v2u*)(&Wl[rbase + c]) = pk.u;
  }
  __syncthreads();
  // O = W @ V over K=NP: 4 interleaved accumulators to break the dependent chain
  const bf16_t* vbase = vt + ((size_t)b * Ec + h * Dc) * NP;
  v8f a0 = {}, a1 = {}, a2 = {}, a3 = {};
  for (int kt = 0; kt < NP / 32; kt += 4) {
    v16bf w0 = load_a_al(Wl, NP, 0, (kt + 0) * 32, lane);
    v16bf v0 = load_bt_al(vbase, NP, (kt + 0) * 32, 0, lane);
    v16bf w1 = load_a_al(Wl, NP, 0, (kt + 1) * 32, lane);
    v16bf v1 = load_bt_al(vbase, NP, (kt + 1) * 32, 0, lane);
    v16bf w2 = load_a_al(Wl, NP, 0, (kt + 2) * 32, lane);
    v16bf v2 = load_bt_al(vbase, NP, (kt + 2) * 32, 0, lane);
    v16bf w3 = load_a_al(Wl, NP, 0, (kt + 3) * 32, lane);
    v16bf v3 = load_bt_al(vbase, NP, (kt + 3) * 32, 0, lane);
    a0 = wmma_bf16(w0, v0, a0);
    a1 = wmma_bf16(w1, v1, a1);
    a2 = wmma_bf16(w2, v2, a2);
    a3 = wmma_bf16(w3, v3, a3);
  }
  v8f acc = (a0 + a1) + (a2 + a3);
  #pragma unroll
  for (int r = 0; r < 8; ++r) {
    int orow = n0 + (hh << 3) + r;
    out[((size_t)b * qpitch + orow) * Ec + h * Dc + nn] = f2bf(acc[r]);
  }
}

// ---------------- decoder prep: gather + concat + residual ----------------
// grid (PPc, Bc), block Ec
__global__ void dec_prep_kernel(const float* __restrict__ xf,
                                const int* __restrict__ current_node,
                                const float* __restrict__ load,
                                const float* __restrict__ cap_w,
                                bf16_t* __restrict__ catb,  // [B,PPc,160] (zero padded)
                                float* __restrict__ resf) { // [B,PPc,128]
  int p = blockIdx.x, b = blockIdx.y, e = threadIdx.x;
  float enc_last = 0.0f, ld = 0.0f;
  if (p < Pc) {
    int cn = current_node[b * Pc + p];
    enc_last = xf[((size_t)b * NP + cn) * Ec + e];
    ld = load[b * Pc + p];
  }
  size_t cbase = ((size_t)b * PPc + p) * 160;
  catb[cbase + e] = f2bf(enc_last);
  if (e < 32) catb[cbase + Ec + e] = (e == 0) ? f2bf(ld) : bfbits(0);
  resf[((size_t)b * PPc + p) * Ec + e] = enc_last + ld * cap_w[e];
}

// ---------------- final epilogue ----------------
__device__ __forceinline__ unsigned hsh(unsigned x) {
  x ^= x >> 16; x *= 0x7feb352du; x ^= x >> 15; x *= 0x846ca68bu; x ^= x >> 16;
  return x;
}
__device__ __forceinline__ float noise_val(int b, int p, int n) {
  unsigned idx = ((unsigned)(b * Pc + p)) * (unsigned)NN + (unsigned)n;
  float u1 = (float)((hsh(idx * 2u + 0x9e3779b9u) >> 8) + 1u) * (1.0f / 16777216.0f);
  float u2 = (float)(hsh(idx * 2u + 1u) >> 8) * (1.0f / 16777216.0f);
  float z = sqrtf(-2.0f * logf(u1)) * __cosf(6.28318530718f * u2);
  return z * 0.2f + 0.5f;
}
__device__ __forceinline__ float blk_red(float v, float* sm, int tid, int op) {
  #pragma unroll
  for (int o = 16; o > 0; o >>= 1) {
    float t = __shfl_xor(v, o, 32);
    v = (op == 0) ? fmaxf(v, t) : (op == 1) ? fminf(v, t) : (v + t);
  }
  __syncthreads();
  if ((tid & 31) == 0) sm[tid >> 5] = v;
  __syncthreads();
  if (tid == 0) {
    float r = sm[0];
    for (int i = 1; i < 8; ++i)
      r = (op == 0) ? fmaxf(r, sm[i]) : (op == 1) ? fminf(r, sm[i]) : (r + sm[i]);
    sm[0] = r;
  }
  __syncthreads();
  return sm[0];
}
// grid (Pc, Bc), block 256
__global__ void final_kernel(const float* __restrict__ score,     // [B,PPc,NP]
                             const float* __restrict__ cur_dist,  // [B,Pc,NN]
                             const float* __restrict__ mask,      // [B,Pc,NN]
                             float* __restrict__ out) {           // [B,Pc,NN]
  __shared__ float sm[8];
  int p = blockIdx.x, b = blockIdx.y, tid = threadIdx.x;
  const float* dr = cur_dist + ((size_t)b * Pc + p) * NN;
  const float* sr = score + ((size_t)b * PPc + p) * NP;
  const float* mr = mask + ((size_t)b * Pc + p) * NN;
  float* orow = out + ((size_t)b * Pc + p) * NN;
  int i1 = tid, i2 = tid + 256;
  float d1 = (i1 < NN) ? dr[i1] : 0.0f;
  float d2 = (i2 < NN) ? dr[i2] : 0.0f;
  float lmin = 1e30f, lmax = -1e30f;
  if (i1 < NN) { lmin = fminf(lmin, d1); lmax = fmaxf(lmax, d1); }
  if (i2 < NN) { lmin = fminf(lmin, d2); lmax = fmaxf(lmax, d2); }
  float dmin = blk_red(lmin, sm, tid, 1);
  float dmax = blk_red(lmax, sm, tid, 0);
  float denom = dmax - dmin + 1e-6f;
  const float nm = 50.0f / (float)NN;
  float s1 = -1e30f, s2 = -1e30f;
  if (i1 < NN) {
    float s = sr[i1] - (1.0f - nm) * logf((d1 - dmin) / denom + 1e-6f) +
              noise_val(b, p, i1) * nm;
    s1 = 10.0f * tanhf(s) + mr[i1];
  }
  if (i2 < NN) {
    float s = sr[i2] - (1.0f - nm) * logf((d2 - dmin) / denom + 1e-6f) +
              noise_val(b, p, i2) * nm;
    s2 = 10.0f * tanhf(s) + mr[i2];
  }
  float smax = blk_red(fmaxf(s1, s2), sm, tid, 0);
  float e1 = (i1 < NN) ? __expf(s1 - smax) : 0.0f;
  float e2 = (i2 < NN) ? __expf(s2 - smax) : 0.0f;
  float ssum = blk_red(e1 + e2, sm, tid, 2);
  float inv = 1.0f / ssum;
  if (i1 < NN) orow[i1] = e1 * inv;
  if (i2 < NN) orow[i2] = e2 * inv;
}

// ---------------- host launchers ----------------
static inline void launch_gemm(hipStream_t s, const bf16_t* A, int lda,
                               const bf16_t* Wp, int Mtiles, int Ntiles, int K,
                               const float* bias, const float* res, int ldr,
                               float* outf, bf16_t* outb, int ldo,
                               float scale, int relu,
                               bf16_t* outbT = nullptr, int tShift = 0, int tCols = 0) {
  int colGroups = Ntiles / 4;
  dim3 g(Mtiles, (colGroups + 3) / 4), blk(32, 4);
  gemm_kernel<<<g, blk, 0, s>>>(A, lda, Wp, K, Ntiles, bias, res, ldr,
                                outf, outb, ldo, outbT, tShift, tCols, scale, relu);
}
static inline void launch_pack(hipStream_t s, const float* src, bf16_t* dst,
                               int K, int N) {
  dim3 g((K + 31) / 32, N / 16), blk(16, 32);
  pack_b_kernel<<<g, blk, 0, s>>>(src, dst, K, N);
}

extern "C" void kernel_launch(void* const* d_in, const int* in_sizes, int n_in,
                              void* d_out, int out_size, void* d_ws, size_t ws_size,
                              hipStream_t stream) {
  // ---- inputs (setup_inputs dict order) ----
  const float* depot_xy     = (const float*)d_in[0];
  const float* node_xy      = (const float*)d_in[1];
  const float* node_demand  = (const float*)d_in[2];
  const float* load_in      = (const float*)d_in[3];
  const float* cur_dist     = (const float*)d_in[4];
  const float* ninf_mask    = (const float*)d_in[5];
  const int*   current_node = (const int*)d_in[6];
  const float* emb_depot_w  = (const float*)d_in[7];
  const float* emb_depot_b  = (const float*)d_in[8];
  const float* emb_node_w   = (const float*)d_in[9];
  const float* emb_node_b   = (const float*)d_in[10];
  const float* enc_wq       = (const float*)d_in[11];
  const float* enc_wk       = (const float*)d_in[12];
  const float* enc_wv       = (const float*)d_in[13];
  const float* enc_comb_w   = (const float*)d_in[14];
  const float* enc_comb_b   = (const float*)d_in[15];
  const float* enc_ff1_w    = (const float*)d_in[16];
  const float* enc_ff1_b    = (const float*)d_in[17];
  const float* enc_ff2_w    = (const float*)d_in[18];
  const float* enc_ff2_b    = (const float*)d_in[19];
  const float* dec_wq_last  = (const float*)d_in[20];
  const float* dec_wk       = (const float*)d_in[21];
  const float* dec_wv       = (const float*)d_in[22];
  const float* dec_comb_w   = (const float*)d_in[23];
  const float* dec_comb_b   = (const float*)d_in[24];
  const float* dec_cap_w    = (const float*)d_in[25];
  const float* dec_ff1_w    = (const float*)d_in[26];
  const float* dec_ff1_b    = (const float*)d_in[27];
  const float* dec_ff2_w    = (const float*)d_in[28];
  const float* dec_ff2_b    = (const float*)d_in[29];
  float* out = (float*)d_out;

  // ---- workspace carve ----
  char* ws = (char*)d_ws;
  size_t off = 0;
  auto carve = [&](size_t bytes) -> char* {
    char* p = ws + off;
    off = (off + bytes + 255) & ~(size_t)255;
    return p;
  };
  const size_t ACT = (size_t)Bc * NP * Ec;   // 16*512*128
  float*  xf   = (float*)carve(ACT * 4);
  bf16_t* xb   = (bf16_t*)carve(ACT * 2);
  bf16_t* qb   = (bf16_t*)carve(ACT * 2);
  bf16_t* kb   = (bf16_t*)carve(ACT * 2);
  bf16_t* vtb  = (bf16_t*)carve(ACT * 2);   // V transposed [B,128,NP]
  bf16_t* mhb  = (bf16_t*)carve(ACT * 2);
  float*  o1f  = (float*)carve(ACT * 4);
  bf16_t* o1b  = (bf16_t*)carve(ACT * 2);
  bf16_t* h1b  = (bf16_t*)carve((size_t)Bc * NP * Fc * 2);
  // packed bf16 weights (B-fragment order, nt-major, per layer)
  bf16_t* wqp   = (bf16_t*)carve((size_t)Lc * Ec * Ec * 2);
  bf16_t* wkp   = (bf16_t*)carve((size_t)Lc * Ec * Ec * 2);
  bf16_t* wvp   = (bf16_t*)carve((size_t)Lc * Ec * Ec * 2);
  bf16_t* wcp   = (bf16_t*)carve((size_t)Lc * Ec * Ec * 2);
  bf16_t* wf1p  = (bf16_t*)carve((size_t)Lc * Ec * Fc * 2);
  bf16_t* wf2p  = (bf16_t*)carve((size_t)Lc * Fc * Ec * 2);
  bf16_t* dwqp  = (bf16_t*)carve((size_t)160 * Ec * 2);   // K padded 129->160
  bf16_t* dwkp  = (bf16_t*)carve((size_t)Ec * Ec * 2);
  bf16_t* dwvp  = (bf16_t*)carve((size_t)Ec * Ec * 2);
  bf16_t* dwcp  = (bf16_t*)carve((size_t)Ec * Ec * 2);
  bf16_t* dwf1p = (bf16_t*)carve((size_t)Ec * Fc * 2);
  bf16_t* dwf2p = (bf16_t*)carve((size_t)Fc * Ec * 2);
  // decoder buffers
  bf16_t* kdb    = (bf16_t*)carve(ACT * 2);
  bf16_t* vdtb   = (bf16_t*)carve(ACT * 2);  // decoder V transposed [B,128,NP]
  bf16_t* catb   = (bf16_t*)carve((size_t)Bc * PPc * 160 * 2);
  float*  resf   = (float*)carve((size_t)Bc * PPc * Ec * 4);
  bf16_t* qdb    = (bf16_t*)carve((size_t)Bc * PPc * Ec * 2);
  bf16_t* mhdb   = (bf16_t*)carve((size_t)Bc * PPc * Ec * 2);
  float*  mh2f   = (float*)carve((size_t)Bc * PPc * Ec * 4);
  bf16_t* mh2b   = (bf16_t*)carve((size_t)Bc * PPc * Ec * 2);
  bf16_t* h1db   = (bf16_t*)carve((size_t)Bc * PPc * Fc * 2);
  bf16_t* qrefb  = (bf16_t*)carve((size_t)Bc * PPc * Ec * 2);
  float*  scoref = (float*)carve((size_t)Bc * PPc * NP * 4);
  (void)ws_size; (void)in_sizes; (void)n_in; (void)out_size;

  // ---- pack weights per layer (nt-major within each layer slice) ----
  for (int l = 0; l < Lc; ++l) {
    launch_pack(stream, enc_wq     + (size_t)l * Ec * Ec, wqp  + (size_t)l * Ec * Ec, Ec, Ec);
    launch_pack(stream, enc_wk     + (size_t)l * Ec * Ec, wkp  + (size_t)l * Ec * Ec, Ec, Ec);
    launch_pack(stream, enc_wv     + (size_t)l * Ec * Ec, wvp  + (size_t)l * Ec * Ec, Ec, Ec);
    launch_pack(stream, enc_comb_w + (size_t)l * Ec * Ec, wcp  + (size_t)l * Ec * Ec, Ec, Ec);
    launch_pack(stream, enc_ff1_w  + (size_t)l * Ec * Fc, wf1p + (size_t)l * Ec * Fc, Ec, Fc);
    launch_pack(stream, enc_ff2_w  + (size_t)l * Fc * Ec, wf2p + (size_t)l * Fc * Ec, Fc, Ec);
  }
  launch_pack(stream, dec_wq_last, dwqp,  129, Ec);
  launch_pack(stream, dec_wk,      dwkp,  Ec,  Ec);
  launch_pack(stream, dec_wv,      dwvp,  Ec,  Ec);
  launch_pack(stream, dec_comb_w,  dwcp,  Ec,  Ec);
  launch_pack(stream, dec_ff1_w,   dwf1p, Ec,  Fc);
  launch_pack(stream, dec_ff2_w,   dwf2p, Fc,  Ec);

  // ---- embedding ----
  embed_kernel<<<dim3(NP, Bc), dim3(Ec), 0, stream>>>(
      depot_xy, node_xy, node_demand, emb_depot_w, emb_depot_b,
      emb_node_w, emb_node_b, xf, xb);

  const int MT = Bc * NP / 16;  // 512 row tiles (batch folded into M)
  // ---- encoder layers ----
  for (int l = 0; l < Lc; ++l) {
    launch_gemm(stream, xb, Ec, wqp + (size_t)l * Ec * Ec, MT, Ec / 16, Ec,
                nullptr, nullptr, 0, nullptr, qb, Ec, 1.0f, 0);
    launch_gemm(stream, xb, Ec, wkp + (size_t)l * Ec * Ec, MT, Ec / 16, Ec,
                nullptr, nullptr, 0, nullptr, kb, Ec, 1.0f, 0);
    launch_gemm(stream, xb, Ec, wvp + (size_t)l * Ec * Ec, MT, Ec / 16, Ec,
                nullptr, nullptr, 0, nullptr, nullptr, Ec, 1.0f, 0,
                vtb, 9 /*NP=512 rows/batch*/, Ec);
    attn_kernel<<<dim3(NP / 16, Hc, Bc), dim3(32), 0, stream>>>(
        qb, NP, kb, vtb, nullptr, 0, 0, mhb, NN, 0.25f);
    // o1 = x + mh @ comb_w + comb_b
    launch_gemm(stream, mhb, Ec, wcp + (size_t)l * Ec * Ec, MT, Ec / 16, Ec,
                enc_comb_b + l * Ec, xf, Ec, o1f, o1b, Ec, 1.0f, 0);
    // h1 = relu(o1 @ ff1 + b1)
    launch_gemm(stream, o1b, Ec, wf1p + (size_t)l * Ec * Fc, MT, Fc / 16, Ec,
                enc_ff1_b + l * Fc, nullptr, 0, nullptr, h1b, Fc, 1.0f, 1);
    // x = o1 + h1 @ ff2 + b2
    launch_gemm(stream, h1b, Fc, wf2p + (size_t)l * Fc * Ec, MT, Ec / 16, Fc,
                enc_ff2_b + l * Ec, o1f, Ec, xf, xb, Ec, 1.0f, 0);
  }

  // ---- decoder ----
  launch_gemm(stream, xb, Ec, dwkp, MT, Ec / 16, Ec, nullptr, nullptr, 0,
              nullptr, kdb, Ec, 1.0f, 0);
  launch_gemm(stream, xb, Ec, dwvp, MT, Ec / 16, Ec, nullptr, nullptr, 0,
              nullptr, nullptr, Ec, 1.0f, 0, vdtb, 9, Ec);
  dec_prep_kernel<<<dim3(PPc, Bc), dim3(Ec), 0, stream>>>(
      xf, current_node, load_in, dec_cap_w, catb, resf);
  const int MTd = Bc * PPc / 16;  // 112 row tiles
  // q = input_cat @ dec_wq_last  (K padded to 160, zeros in both operands)
  launch_gemm(stream, catb, 160, dwqp, MTd, Ec / 16, 160, nullptr, nullptr, 0,
              nullptr, qdb, Ec, 1.0f, 0);
  attn_kernel<<<dim3(PPc / 16, Hc, Bc), dim3(32), 0, stream>>>(
      qdb, PPc, kdb, vdtb, ninf_mask, Pc, NN, mhdb, NN, 0.25f);
  // mh2 = mh @ comb + comb_b + (encoded_last + load*cap_w)
  launch_gemm(stream, mhdb, Ec, dwcp, MTd, Ec / 16, Ec, dec_comb_b,
              resf, Ec, mh2f, mh2b, Ec, 1.0f, 0);
  // h1 = relu(mh2 @ ff1 + b1)
  launch_gemm(stream, mh2b, Ec, dwf1p, MTd, Fc / 16, Ec, dec_ff1_b,
              nullptr, 0, nullptr, h1db, Fc, 1.0f, 1);
  // q_ref = h1 @ ff2 + b2 + mh2
  launch_gemm(stream, h1db, Fc, dwf2p, MTd, Ec / 16, Fc, dec_ff2_b,
              mh2f, Ec, nullptr, qrefb, Ec, 1.0f, 0);
  // score = q_ref @ encoded^T / sqrt(E)
  gemm_bt_kernel<<<dim3(PPc / 16, NP / 16 / 4, Bc), dim3(32), 0, stream>>>(
      qrefb, Ec, PPc * Ec, xb, Ec, NP * Ec,
      scoref, NP, PPc * NP, Ec, 1.0f / sqrtf((float)Ec));
  // final epilogue + softmax
  final_kernel<<<dim3(Pc, Bc), dim3(256), 0, stream>>>(
      scoref, cur_dist, ninf_mask, out);
}
